// FullAttentionEncoder_24111946400275
// MI455X (gfx1250) — matrix operand order repaired and checked by reading, not compiled, not run on hardware
//
#include <hip/hip_runtime.h>

// ---- CDNA5 (gfx1250) types ----
typedef __bf16 bf16_t;
typedef bf16_t v16bf __attribute__((ext_vector_type(16)));
typedef bf16_t v8bf  __attribute__((ext_vector_type(8)));
typedef float  v8f   __attribute__((ext_vector_type(8)));
typedef float  v4f   __attribute__((ext_vector_type(4)));

#define Bsz 4
#define Lsz 2048
#define Hsz 8
#define Esz 64
#define SCALE 0.125f            // 1/sqrt(64)
#define NEGBIG (-1e30f)
#define NWAVES 8
#define MAXT 64                 // worst-case active key tiles per query tile (15)
#define NKT (Lsz / 16)

static __device__ __forceinline__ v8f wmma_bf16(v16bf a, v16bf b, v8f c) {
  // D = A(16x32 bf16) * B(32x16 bf16) + C(16x16 f32)
  return __builtin_amdgcn_wmma_f32_16x16x32_bf16(false, a, false, b, (short)0, c,
                                                 false, false);
}

// reference mask: |d|<=4 OR |d| is a power of two
static __device__ __forceinline__ bool allowed_d(int d) {
  int ad = d < 0 ? -d : d;
  return (ad <= 4) || ((ad & (ad - 1)) == 0);
}

__global__ __launch_bounds__(256)
void logsparse_attn_kernel(const float* __restrict__ Q,
                           const float* __restrict__ K,
                           const float* __restrict__ V,
                           float* __restrict__ O) {
  __shared__ alignas(64) unsigned short s_list[NWAVES][MAXT];
  __shared__ alignas(64) bf16_t s_P[NWAVES][16 * 32];    // P tile, row-major [m][k]
  __shared__ alignas(64) bf16_t s_VT[NWAVES][Esz * 32];  // V pair tile, [e][k]

  const int lane = threadIdx.x & 31;
  const int wave = threadIdx.x >> 5;
  const int hlf  = lane >> 4;   // which 16-lane half
  const int lr   = lane & 15;

  const int bid  = blockIdx.x;
  const int qblk = bid & 15;            // 16 blocks per (b,h): 16*8 waves*16 rows = 2048
  const int bh   = bid >> 4;
  const int b    = bh >> 3;             // H = 8
  const int h    = bh & 7;
  const int qt   = qblk * NWAVES + wave;
  const int i0   = qt * 16;

  // ---------- Q fragment: A-matrix layout, two K=32 slices over E=64 ----------
  // A 16x32 bf16: lane holds row M=lane%16; element e -> K=hlf*16+e
  const float* qrow = Q + ((size_t)((b * Lsz + i0 + lr) * Hsz + h)) * Esz;
  v16bf aq[2];
#pragma unroll
  for (int es = 0; es < 2; ++es) {
    const float* p0 = qrow + es * 32 + hlf * 16;       // K = hlf*16 + 0..7
    const float* p1 = qrow + es * 32 + hlf * 16 + 8;   // K = hlf*16 + 8..15
    v4f x0 = *(const v4f*)p0, x1 = *(const v4f*)(p0 + 4);
    v4f y0 = *(const v4f*)p1, y1 = *(const v4f*)(p1 + 4);
#pragma unroll
    for (int t = 0; t < 4; ++t) {
      aq[es][t]      = (bf16_t)x0[t];
      aq[es][4 + t]  = (bf16_t)x1[t];
      aq[es][8 + t]  = (bf16_t)y0[t];
      aq[es][12 + t] = (bf16_t)y1[t];
    }
  }

  // ---------- build active key-tile list (wave-private) ----------
  int cnt = 0;
  for (int kt = 0; kt < NKT; ++kt) {
    // d = j - i spans [kt*16 - i0 - 15, kt*16 - i0 + 15]; every d in range occurs
    int d = kt * 16 - i0 - 15 + lane;
    bool ok = (lane < 31) && allowed_d(d);
    if (__ballot(ok)) {                 // wave-uniform
      if (lane == 0) s_list[wave][cnt] = (unsigned short)kt;
      ++cnt;
    }
  }
  asm volatile("s_wait_dscnt 0" ::: "memory");

  // ---------- flash-attention running state ----------
  // C/D layout: lane holds col N=lane%16; VGPR j holds row M=j+8*hlf
  float mrow[8], lrow[8];
  v8f acc[4];
#pragma unroll
  for (int j = 0; j < 8; ++j) { mrow[j] = NEGBIG; lrow[j] = 0.0f; }
#pragma unroll
  for (int t = 0; t < 4; ++t) { v8f z = {}; acc[t] = z; }

  for (int p = 0; p < cnt; p += 2) {
    const int ktA = (int)s_list[wave][p];
    const int ktB = (p + 1 < cnt) ? (int)s_list[wave][p + 1] : -1;

    // --- K fragments (B-matrix, K-dim = feature e): element ei -> e = es*32 + hlf*16 + ei
    v16bf bkA[2], bkB[2];
    {
      const float* krA = K + ((size_t)((b * Lsz + ktA * 16 + lr) * Hsz + h)) * Esz;
#pragma unroll
      for (int es = 0; es < 2; ++es) {
        const float* pp = krA + es * 32 + hlf * 16;
        v4f c0 = *(const v4f*)pp,       c1 = *(const v4f*)(pp + 4);
        v4f c2 = *(const v4f*)(pp + 8), c3 = *(const v4f*)(pp + 12);
#pragma unroll
        for (int t = 0; t < 4; ++t) {
          bkA[es][t]      = (bf16_t)c0[t];
          bkA[es][4 + t]  = (bf16_t)c1[t];
          bkA[es][8 + t]  = (bf16_t)c2[t];
          bkA[es][12 + t] = (bf16_t)c3[t];
        }
      }
      if (ktB >= 0) {
        const float* krB = K + ((size_t)((b * Lsz + ktB * 16 + lr) * Hsz + h)) * Esz;
#pragma unroll
        for (int es = 0; es < 2; ++es) {
          const float* pp = krB + es * 32 + hlf * 16;
          v4f c0 = *(const v4f*)pp,       c1 = *(const v4f*)(pp + 4);
          v4f c2 = *(const v4f*)(pp + 8), c3 = *(const v4f*)(pp + 12);
#pragma unroll
          for (int t = 0; t < 4; ++t) {
            bkB[es][t]      = (bf16_t)c0[t];
            bkB[es][4 + t]  = (bf16_t)c1[t];
            bkB[es][8 + t]  = (bf16_t)c2[t];
            bkB[es][12 + t] = (bf16_t)c3[t];
          }
        }
      } else {
#pragma unroll
        for (int es = 0; es < 2; ++es)
#pragma unroll
          for (int t = 0; t < 16; ++t) bkB[es][t] = (bf16_t)0.0f;
      }
    }

    // --- scores: S = Q * K^T for the two 16x16 tiles (E=64 -> 2 x K=32) ---
    v8f s0 = {}; s0 = wmma_bf16(aq[0], bkA[0], s0); s0 = wmma_bf16(aq[1], bkA[1], s0);
    v8f s1 = {}; s1 = wmma_bf16(aq[0], bkB[0], s1); s1 = wmma_bf16(aq[1], bkB[1], s1);

    // --- stage V pair tile into LDS, transposed [e][k], bf16 ---
    {
      const int ktv = (lane < 16) ? ktA : ktB;
      if (ktv >= 0) {
        const float* vr = V + ((size_t)((b * Lsz + ktv * 16 + lr) * Hsz + h)) * Esz;
#pragma unroll
        for (int e4 = 0; e4 < 16; ++e4) {
          v4f vv = *(const v4f*)(vr + e4 * 4);
#pragma unroll
          for (int t = 0; t < 4; ++t)
            s_VT[wave][(e4 * 4 + t) * 32 + lane] = (bf16_t)vv[t];
        }
      } else {
#pragma unroll
        for (int e = 0; e < Esz; ++e) s_VT[wave][e * 32 + lane] = (bf16_t)0.0f;
      }
    }

    // --- mask + scale + online softmax (per-row over this lane's 8 rows) ---
    const int jAg = ktA * 16 + lr;
    const int jBg = (ktB >= 0) ? (ktB * 16 + lr) : 0;
#pragma unroll
    for (int j = 0; j < 8; ++j) {
      const int ig = i0 + j + hlf * 8;
      float v0 = allowed_d(jAg - ig) ? s0[j] * SCALE : NEGBIG;
      float v1 = (ktB >= 0 && allowed_d(jBg - ig)) ? s1[j] * SCALE : NEGBIG;
      float rm = fmaxf(v0, v1);
#pragma unroll
      for (int off = 1; off < 16; off <<= 1)
        rm = fmaxf(rm, __shfl_xor(rm, off, 16));
      const float mo = mrow[j];
      const float mn = fmaxf(mo, rm);
      const float r  = __expf(mo - mn);   // exp(-1e30-x) underflows to exactly 0
      const float p0 = __expf(v0 - mn);
      const float p1 = __expf(v1 - mn);
      float rs = p0 + p1;
#pragma unroll
      for (int off = 1; off < 16; off <<= 1)
        rs += __shfl_xor(rs, off, 16);
      mrow[j] = mn;
      lrow[j] = lrow[j] * r + rs;
#pragma unroll
      for (int t = 0; t < 4; ++t) acc[t][j] *= r;
      const int pm = j + hlf * 8;                       // P row (query)
      s_P[wave][pm * 32 + lr]      = (bf16_t)p0;        // keys of tile A -> K 0..15
      s_P[wave][pm * 32 + 16 + lr] = (bf16_t)p1;        // keys of tile B -> K 16..31
    }

    asm volatile("s_wait_dscnt 0" ::: "memory");

    // --- P·V: A = P (16x32), B = V (32x16 per E-tile), 4 output tiles over E=64 ---
    v16bf ap;
    {
      const bf16_t* pr = &s_P[wave][lr * 32];
      v8bf lo = *(const v8bf*)(pr + hlf * 16);
      v8bf hi = *(const v8bf*)(pr + hlf * 16 + 8);
#pragma unroll
      for (int t = 0; t < 8; ++t) { ap[t] = lo[t]; ap[8 + t] = hi[t]; }
    }
#pragma unroll
    for (int t = 0; t < 4; ++t) {
      const bf16_t* vp = &s_VT[wave][(t * 16 + lr) * 32 + hlf * 16];
      v16bf bv = *(const v16bf*)vp;     // 16 consecutive K values, 32B aligned
      acc[t] = wmma_bf16(ap, bv, acc[t]);
    }
    // compiler barrier: keep next iteration's LDS stores after these loads
    asm volatile("" ::: "memory");
  }

  // ---------- epilogue: normalize by row sums and store ----------
#pragma unroll
  for (int j = 0; j < 8; ++j) {
    const float inv = 1.0f / lrow[j];   // every row has its diagonal -> lrow > 0
    const int ig = i0 + j + hlf * 8;
    float* orow = O + ((size_t)((b * Lsz + ig) * Hsz + h)) * Esz;
#pragma unroll
    for (int t = 0; t < 4; ++t)
      orow[t * 16 + lr] = acc[t][j] * inv;
  }
}

extern "C" void kernel_launch(void* const* d_in, const int* in_sizes, int n_in,
                              void* d_out, int out_size, void* d_ws, size_t ws_size,
                              hipStream_t stream) {
  (void)in_sizes; (void)n_in; (void)d_ws; (void)ws_size; (void)out_size;
  const float* Q = (const float*)d_in[0];
  const float* K = (const float*)d_in[1];
  const float* V = (const float*)d_in[2];
  // d_in[3] is the boolean mask; the kernel regenerates it analytically.
  float* O = (float*)d_out;
  dim3 grid(16 * Bsz * Hsz);   // 512 blocks: 8 query tiles (waves) each
  dim3 block(256);             // 8 wave32
  logsparse_attn_kernel<<<grid, block, 0, stream>>>(Q, K, V, O);
}
